// VehicleTrajectoryDecoder_87952340287787
// MI455X (gfx1250) — matrix operand order repaired and checked by reading, not compiled, not run on hardware
//
#include <hip/hip_runtime.h>

typedef _Float16 v16h __attribute__((ext_vector_type(16)));
typedef _Float16 v8h  __attribute__((ext_vector_type(8)));
typedef float    v8f  __attribute__((ext_vector_type(8)));
typedef float    v4f  __attribute__((ext_vector_type(4)));

#define B_   32
#define N_   256
#define D_   512
#define H_   8
#define T_   60
#define DFF_ 2048
#define BD_  (B_ * D_)

// ---------------------------------------------------------------------------
// WMMA GEMM: C[M x N] = act( A[MxK](f32, row-major) @ W[N x K](f16, row-major)^T + bias )
// One wave (32 threads) computes a 32(M) x 64(N) tile, K stepped by 32.
// Each B fragment is reused for two M-tiles: 8 WMMAs per 16 b128 loads.
// ---------------------------------------------------------------------------
__global__ __launch_bounds__(32) void gemm_xwT(
    const float* __restrict__ A, const _Float16* __restrict__ W,
    const float* __restrict__ bias, float* __restrict__ C,
    int M, int N, int K, int relu)
{
    const int lane = threadIdx.x;
    const int l16  = lane & 15;
    const int half = lane >> 4;
    const int n0   = blockIdx.x * 64;
    const int m0   = blockIdx.y * 32;

    v8f zero;
    #pragma unroll
    for (int i = 0; i < 8; ++i) zero[i] = 0.0f;
    v8f acc[2][4];
    #pragma unroll
    for (int s = 0; s < 4; ++s) { acc[0][s] = zero; acc[1][s] = zero; }

    const float* arow0 = A + (size_t)(m0 + l16) * K;
    const float* arow1 = A + (size_t)(m0 + 16 + l16) * K;

    for (int k0 = 0; k0 < K; k0 += 32) {
        // A fragments: lane half h covers K = h*8+{0..7} and 16+h*8+{0..7}
        v16h af0, af1;
        {
            const v4f a0 = *(const v4f*)(arow0 + k0 + half * 8);
            const v4f a1 = *(const v4f*)(arow0 + k0 + half * 8 + 4);
            const v4f a2 = *(const v4f*)(arow0 + k0 + 16 + half * 8);
            const v4f a3 = *(const v4f*)(arow0 + k0 + 16 + half * 8 + 4);
            #pragma unroll
            for (int i = 0; i < 4; ++i) {
                af0[i]      = (_Float16)a0[i];
                af0[4 + i]  = (_Float16)a1[i];
                af0[8 + i]  = (_Float16)a2[i];
                af0[12 + i] = (_Float16)a3[i];
            }
        }
        {
            const v4f a0 = *(const v4f*)(arow1 + k0 + half * 8);
            const v4f a1 = *(const v4f*)(arow1 + k0 + half * 8 + 4);
            const v4f a2 = *(const v4f*)(arow1 + k0 + 16 + half * 8);
            const v4f a3 = *(const v4f*)(arow1 + k0 + 16 + half * 8 + 4);
            #pragma unroll
            for (int i = 0; i < 4; ++i) {
                af1[i]      = (_Float16)a0[i];
                af1[4 + i]  = (_Float16)a1[i];
                af1[8 + i]  = (_Float16)a2[i];
                af1[12 + i] = (_Float16)a3[i];
            }
        }
        #pragma unroll
        for (int s = 0; s < 4; ++s) {
            // B fragment: lane half h covers K = h*8+{0..7} and 16+h*8+{0..7}, N = n0+s*16+l16
            const _Float16* wp = W + (size_t)(n0 + s * 16 + l16) * K + k0 + half * 8;
            const v8h b0 = *(const v8h*)(wp);
            const v8h b1 = *(const v8h*)(wp + 16);
            v16h bf;
            #pragma unroll
            for (int i = 0; i < 8; ++i) { bf[i] = b0[i]; bf[8 + i] = b1[i]; }
            acc[0][s] = __builtin_amdgcn_wmma_f32_16x16x32_f16(
                false, af0, false, bf, (short)0, acc[0][s], false, false);
            acc[1][s] = __builtin_amdgcn_wmma_f32_16x16x32_f16(
                false, af1, false, bf, (short)0, acc[1][s], false, false);
        }
    }

    // Epilogue: lane l, VGPR r -> M = mtile*16 + (l/16)*8 + r, N = l%16
    #pragma unroll
    for (int mt = 0; mt < 2; ++mt) {
        const int mb = m0 + mt * 16 + half * 8;
        #pragma unroll
        for (int s = 0; s < 4; ++s) {
            const int col = n0 + s * 16 + l16;
            const float bv = bias ? bias[col] : 0.0f;
            #pragma unroll
            for (int r = 0; r < 8; ++r) {
                float v = acc[mt][s][r] + bv;
                if (relu) v = v > 0.0f ? v : 0.0f;
                C[(size_t)(mb + r) * N + col] = v;
            }
        }
    }
}

// ---------------------------------------------------------------------------
// Small helper kernels
// ---------------------------------------------------------------------------
__global__ void cvt_f16_k(const float* __restrict__ s, _Float16* __restrict__ d, int n) {
    int i = blockIdx.x * blockDim.x + threadIdx.x;
    if (i < n) d[i] = (_Float16)s[i];
}

__global__ void copy_f32_k(const float* __restrict__ s, float* __restrict__ d, int n) {
    int i = blockIdx.x * blockDim.x + threadIdx.x;
    if (i < n) d[i] = s[i];
}

__global__ void prep_x_k(const float* __restrict__ tok_t, const float* __restrict__ pe_t,
                         float* __restrict__ xbuf) {
    int i = blockIdx.x * blockDim.x + threadIdx.x; // 0 .. BD_-1
    xbuf[i] = tok_t[i] + pe_t[i & (D_ - 1)];
}

__global__ void append_kv_k(const float* __restrict__ proj, float* __restrict__ dK,
                            float* __restrict__ dV) {
    int i = blockIdx.x * blockDim.x + threadIdx.x; // 0 .. BD_-1
    int b = i >> 9, d = i & (D_ - 1);
    dK[i] = proj[(size_t)b * 2048 + 1024 + d];
    dV[i] = proj[(size_t)b * 2048 + 1536 + d];
}

__global__ void finish_step_k(const float* __restrict__ x3, const float* __restrict__ ctx,
                              float* __restrict__ tok_next, float* __restrict__ outseq, int t) {
    int i = blockIdx.x * blockDim.x + threadIdx.x; // 0 .. BD_-1
    int b = i >> 9, d = i & (D_ - 1);
    float v = x3[i] + ctx[i];
    tok_next[i] = v;
    outseq[((size_t)b * T_ + t) * D_ + d] = v;
}

// ---------------------------------------------------------------------------
// Residual + LayerNorm: O = LN(X + R) * g + b   (row = one batch element, D=512)
// ---------------------------------------------------------------------------
__global__ __launch_bounds__(256) void resid_ln_k(
    const float* __restrict__ X, const float* __restrict__ R,
    const float* __restrict__ g, const float* __restrict__ be,
    float* __restrict__ O)
{
    __shared__ float red[256];
    const int b = blockIdx.x, tid = threadIdx.x;
    float v0 = X[(size_t)b * D_ + tid]       + R[(size_t)b * D_ + tid];
    float v1 = X[(size_t)b * D_ + 256 + tid] + R[(size_t)b * D_ + 256 + tid];
    red[tid] = v0 + v1;
    __syncthreads();
    for (int o = 128; o > 0; o >>= 1) { if (tid < o) red[tid] += red[tid + o]; __syncthreads(); }
    const float mean = red[0] * (1.0f / (float)D_);
    __syncthreads();
    const float d0 = v0 - mean, d1 = v1 - mean;
    red[tid] = d0 * d0 + d1 * d1;
    __syncthreads();
    for (int o = 128; o > 0; o >>= 1) { if (tid < o) red[tid] += red[tid + o]; __syncthreads(); }
    const float rstd = rsqrtf(red[0] * (1.0f / (float)D_) + 1e-5f);
    O[(size_t)b * D_ + tid]       = d0 * rstd * g[tid]       + be[tid];
    O[(size_t)b * D_ + 256 + tid] = d1 * rstd * g[256 + tid] + be[256 + tid];
}

// ---------------------------------------------------------------------------
// Attention, 1 query x 256 keys, dh=64, per (b,h) block of 256 threads.
// K/V layout: (B*N, D) with heads packed along D.
// ---------------------------------------------------------------------------
__global__ __launch_bounds__(256) void attn256_k(
    const float* __restrict__ Q, int qstride, int qoff,
    const float* __restrict__ Kc, const float* __restrict__ Vc,
    float* __restrict__ O)
{
    const int b = blockIdx.x, h = blockIdx.y, tid = threadIdx.x;
    __shared__ float qs[64];
    __shared__ float pp[256];
    __shared__ float red[256];
    if (tid < 64) qs[tid] = Q[(size_t)b * qstride + qoff + h * 64 + tid];
    __syncthreads();
    const float* kp = Kc + ((size_t)(b * N_ + tid)) * D_ + h * 64;
    float s = 0.0f;
    #pragma unroll 8
    for (int d = 0; d < 64; ++d) s += qs[d] * kp[d];
    s *= 0.125f; // 1/sqrt(64)
    red[tid] = s;
    __syncthreads();
    for (int o = 128; o > 0; o >>= 1) { if (tid < o) red[tid] = fmaxf(red[tid], red[tid + o]); __syncthreads(); }
    const float mx = red[0];
    __syncthreads();
    const float e = expf(s - mx);
    pp[tid] = e;
    red[tid] = e;
    __syncthreads();
    for (int o = 128; o > 0; o >>= 1) { if (tid < o) red[tid] += red[tid + o]; __syncthreads(); }
    const float inv = 1.0f / red[0];
    if (tid < 64) {
        const float* vp = Vc + (size_t)b * N_ * D_ + h * 64 + tid;
        float o = 0.0f;
        for (int k = 0; k < N_; ++k) o += pp[k] * vp[(size_t)k * D_];
        O[(size_t)b * D_ + h * 64 + tid] = o * inv;
    }
}

// ---------------------------------------------------------------------------
// Incremental self-attention: 1 query x tlen cached keys (tlen <= 60).
// Q from proj_out columns [512,1024). Cache layout: (T, B, D).
// ---------------------------------------------------------------------------
__global__ __launch_bounds__(64) void self_attn_k(
    const float* __restrict__ Q, const float* __restrict__ Ksa,
    const float* __restrict__ Vsa, float* __restrict__ O, int tlen)
{
    const int b = blockIdx.x, h = blockIdx.y, tid = threadIdx.x;
    __shared__ float qs[64];
    __shared__ float pp[64];
    __shared__ float red[64];
    qs[tid] = Q[(size_t)b * 2048 + 512 + h * 64 + tid];
    __syncthreads();
    float s = -1e30f;
    if (tid < tlen) {
        const float* kp = Ksa + ((size_t)tid * B_ + b) * D_ + h * 64;
        float acc = 0.0f;
        #pragma unroll 8
        for (int d = 0; d < 64; ++d) acc += qs[d] * kp[d];
        s = acc * 0.125f;
    }
    red[tid] = s;
    __syncthreads();
    for (int o = 32; o > 0; o >>= 1) { if (tid < o) red[tid] = fmaxf(red[tid], red[tid + o]); __syncthreads(); }
    const float mx = red[0];
    __syncthreads();
    const float e = (tid < tlen) ? expf(s - mx) : 0.0f;
    pp[tid] = e;
    red[tid] = e;
    __syncthreads();
    for (int o = 32; o > 0; o >>= 1) { if (tid < o) red[tid] += red[tid + o]; __syncthreads(); }
    const float inv = 1.0f / red[0];
    const float* vp = Vsa + (size_t)b * D_ + h * 64 + tid;
    float o = 0.0f;
    for (int j = 0; j < tlen; ++j) o += pp[j] * vp[(size_t)j * BD_];
    O[(size_t)b * D_ + h * 64 + tid] = o * inv;
}

// ---------------------------------------------------------------------------
// Final projection: out[b,t,:2] = outseq[b,t,:] @ Wout^T + bout
// ---------------------------------------------------------------------------
__global__ __launch_bounds__(64) void out_proj_k(
    const float* __restrict__ seq, const float* __restrict__ Wout,
    const float* __restrict__ bout, float* __restrict__ out)
{
    const int bt = blockIdx.x, tid = threadIdx.x;
    __shared__ float red[64];
    const float* x = seq + (size_t)bt * D_;
    for (int o = 0; o < 2; ++o) {
        float acc = 0.0f;
        for (int d = tid; d < D_; d += 64) acc += x[d] * Wout[(size_t)o * D_ + d];
        red[tid] = acc;
        __syncthreads();
        for (int w = 32; w > 0; w >>= 1) { if (tid < w) red[tid] += red[tid + w]; __syncthreads(); }
        if (tid == 0) out[(size_t)bt * 2 + o] = red[0] + bout[o];
        __syncthreads();
    }
}

// ---------------------------------------------------------------------------
// Host orchestration
// ---------------------------------------------------------------------------
extern "C" void kernel_launch(void* const* d_in, const int* in_sizes, int n_in,
                              void* d_out, int out_size, void* d_ws, size_t ws_size,
                              hipStream_t stream)
{
    (void)in_sizes; (void)n_in; (void)out_size; (void)ws_size;
    const float* Hv    = (const float*)d_in[0];
    const float* Hv0   = (const float*)d_in[1];
    const float* pe    = (const float*)d_in[2];
    const float* s_Wq  = (const float*)d_in[3];
    const float* s_Wk  = (const float*)d_in[4];
    const float* s_Wv  = (const float*)d_in[5];
    const float* s_Wo  = (const float*)d_in[6];
    const float* s_bq  = (const float*)d_in[7];
    const float* s_bk  = (const float*)d_in[8];
    const float* s_bv  = (const float*)d_in[9];
    const float* s_bo  = (const float*)d_in[10];
    const float* sa_Wq = (const float*)d_in[11];
    const float* sa_Wk = (const float*)d_in[12];
    const float* sa_Wv = (const float*)d_in[13];
    const float* sa_Wo = (const float*)d_in[14];
    const float* sa_bq = (const float*)d_in[15];
    const float* sa_bk = (const float*)d_in[16];
    const float* sa_bv = (const float*)d_in[17];
    const float* sa_bo = (const float*)d_in[18];
    const float* ca_Wq = (const float*)d_in[19];
    const float* ca_Wk = (const float*)d_in[20];
    const float* ca_Wv = (const float*)d_in[21];
    const float* ca_Wo = (const float*)d_in[22];
    const float* ca_bq = (const float*)d_in[23];
    const float* ca_bk = (const float*)d_in[24];
    const float* ca_bv = (const float*)d_in[25];
    const float* ca_bo = (const float*)d_in[26];
    const float* ln1w  = (const float*)d_in[27];
    const float* ln1b  = (const float*)d_in[28];
    const float* ln2w  = (const float*)d_in[29];
    const float* ln2b  = (const float*)d_in[30];
    const float* ln3w  = (const float*)d_in[31];
    const float* ln3b  = (const float*)d_in[32];
    const float* W1    = (const float*)d_in[33];
    const float* b1    = (const float*)d_in[34];
    const float* W2    = (const float*)d_in[35];
    const float* b2    = (const float*)d_in[36];
    const float* Wout  = (const float*)d_in[37];
    const float* bout  = (const float*)d_in[38];

    // --- workspace carve-out ---
    size_t off = 0;
    auto alloc = [&](size_t bytes) -> void* {
        off = (off + 255) & ~(size_t)255;
        void* p = (char*)d_ws + off;
        off += bytes;
        return p;
    };
    const size_t DDh = (size_t)D_ * D_ * sizeof(_Float16);
    _Float16* Wqkv4 = (_Float16*)alloc(4 * DDh);        // [s_Wq; sa_Wq; sa_Wk; sa_Wv]
    _Float16* sWk_h  = (_Float16*)alloc(DDh);
    _Float16* sWv_h  = (_Float16*)alloc(DDh);
    _Float16* caWk_h = (_Float16*)alloc(DDh);
    _Float16* caWv_h = (_Float16*)alloc(DDh);
    _Float16* sWo_h  = (_Float16*)alloc(DDh);
    _Float16* saWo_h = (_Float16*)alloc(DDh);
    _Float16* caWq_h = (_Float16*)alloc(DDh);
    _Float16* caWo_h = (_Float16*)alloc(DDh);
    _Float16* W1_h   = (_Float16*)alloc((size_t)DFF_ * D_ * sizeof(_Float16));
    _Float16* W2_h   = (_Float16*)alloc((size_t)D_ * DFF_ * sizeof(_Float16));
    float* bias4  = (float*)alloc(2048 * sizeof(float));
    float* Ks     = (float*)alloc((size_t)B_ * N_ * D_ * sizeof(float));
    float* Vs     = (float*)alloc((size_t)B_ * N_ * D_ * sizeof(float));
    float* Kc     = (float*)alloc((size_t)B_ * N_ * D_ * sizeof(float));
    float* Vc     = (float*)alloc((size_t)B_ * N_ * D_ * sizeof(float));
    float* saK    = (float*)alloc((size_t)T_ * BD_ * sizeof(float));
    float* saV    = (float*)alloc((size_t)T_ * BD_ * sizeof(float));
    float* tok    = (float*)alloc((size_t)(T_ + 1) * BD_ * sizeof(float));
    float* outseq = (float*)alloc((size_t)B_ * T_ * D_ * sizeof(float));
    float* xbuf   = (float*)alloc(BD_ * sizeof(float));
    float* proj   = (float*)alloc((size_t)B_ * 2048 * sizeof(float));
    float* ctxpre = (float*)alloc(BD_ * sizeof(float));
    float* saout  = (float*)alloc(BD_ * sizeof(float));
    float* x1     = (float*)alloc(BD_ * sizeof(float));
    float* qc     = (float*)alloc(BD_ * sizeof(float));
    float* capre  = (float*)alloc(BD_ * sizeof(float));
    float* x2     = (float*)alloc(BD_ * sizeof(float));
    float* x3     = (float*)alloc(BD_ * sizeof(float));
    float* tmp    = (float*)alloc(BD_ * sizeof(float));
    float* ctxb   = (float*)alloc(BD_ * sizeof(float));
    float* hb     = (float*)alloc((size_t)B_ * DFF_ * sizeof(float));

    auto cvt = [&](const float* s, _Float16* d, int n) {
        cvt_f16_k<<<dim3((n + 255) / 256), 256, 0, stream>>>(s, d, n);
    };
    auto cpy = [&](const float* s, float* d, int n) {
        copy_f32_k<<<dim3((n + 255) / 256), 256, 0, stream>>>(s, d, n);
    };
    auto gemm = [&](const float* A, const _Float16* W, const float* bias, float* C,
                    int M, int N, int K, int relu) {
        gemm_xwT<<<dim3(N / 64, M / 32), 32, 0, stream>>>(A, W, bias, C, M, N, K, relu);
    };

    // --- one-time precompute ---
    const int DD = D_ * D_;
    cvt(s_Wq,  Wqkv4,          DD);
    cvt(sa_Wq, Wqkv4 + DD,     DD);
    cvt(sa_Wk, Wqkv4 + 2 * DD, DD);
    cvt(sa_Wv, Wqkv4 + 3 * DD, DD);
    cpy(s_bq,  bias4,        D_);
    cpy(sa_bq, bias4 + 512,  D_);
    cpy(sa_bk, bias4 + 1024, D_);
    cpy(sa_bv, bias4 + 1536, D_);
    cvt(s_Wk,  sWk_h,  DD);  cvt(s_Wv,  sWv_h,  DD);
    cvt(ca_Wk, caWk_h, DD);  cvt(ca_Wv, caWv_h, DD);
    cvt(s_Wo,  sWo_h,  DD);  cvt(sa_Wo, saWo_h, DD);
    cvt(ca_Wq, caWq_h, DD);  cvt(ca_Wo, caWo_h, DD);
    cvt(W1, W1_h, DFF_ * D_);
    cvt(W2, W2_h, D_ * DFF_);
    cpy(Hv0, tok, BD_); // tgt position 0 = H_v0_init

    // Memory K/V projections (K/V stay resident in 192MB L2 for the whole loop)
    gemm(Hv, sWk_h,  s_bk,  Ks, B_ * N_, D_, D_, 0);
    gemm(Hv, sWv_h,  s_bv,  Vs, B_ * N_, D_, D_, 0);
    gemm(Hv, caWk_h, ca_bk, Kc, B_ * N_, D_, D_, 0);
    gemm(Hv, caWv_h, ca_bv, Vc, B_ * N_, D_, D_, 0);

    // --- autoregressive decode, incremental KV cache ---
    for (int t = 0; t < T_; ++t) {
        prep_x_k<<<BD_ / 256, 256, 0, stream>>>(tok + (size_t)t * BD_, pe + (size_t)t * D_, xbuf);
        // fused {s_q | sa_q | sa_k | sa_v} projection of the new token
        gemm(xbuf, Wqkv4, bias4, proj, B_, 2048, D_, 0);
        append_kv_k<<<BD_ / 256, 256, 0, stream>>>(proj, saK + (size_t)t * BD_, saV + (size_t)t * BD_);
        // spatial attention (ctx), query = current token
        attn256_k<<<dim3(B_, H_), 256, 0, stream>>>(proj, 2048, 0, Ks, Vs, ctxpre);
        // decoder self-attention over cached t+1 tokens
        self_attn_k<<<dim3(B_, H_), 64, 0, stream>>>(proj, saK, saV, saout, t + 1);
        gemm(saout, saWo_h, sa_bo, tmp, B_, D_, D_, 0);
        resid_ln_k<<<B_, 256, 0, stream>>>(tmp, xbuf, ln1w, ln1b, x1);
        // cross-attention to memory
        gemm(x1, caWq_h, ca_bq, qc, B_, D_, D_, 0);
        attn256_k<<<dim3(B_, H_), 256, 0, stream>>>(qc, D_, 0, Kc, Vc, capre);
        gemm(capre, caWo_h, ca_bo, tmp, B_, D_, D_, 0);
        resid_ln_k<<<B_, 256, 0, stream>>>(tmp, x1, ln2w, ln2b, x2);
        // FFN
        gemm(x2, W1_h, b1, hb, B_, DFF_, D_, 1);
        gemm(hb, W2_h, b2, tmp, B_, D_, DFF_, 0);
        resid_ln_k<<<B_, 256, 0, stream>>>(tmp, x2, ln3w, ln3b, x3);
        // ctx output projection + next token
        gemm(ctxpre, sWo_h, s_bo, ctxb, B_, D_, D_, 0);
        finish_step_k<<<BD_ / 256, 256, 0, stream>>>(x3, ctxb, tok + (size_t)(t + 1) * BD_, outseq, t);
    }

    out_proj_k<<<B_ * T_, 64, 0, stream>>>(outseq, Wout, bout, (float*)d_out);
}
